// GptOssDecoderLayer_11424613007753
// MI455X (gfx1250) — matrix operand order repaired and checked by reading, not compiled, not run on hardware
//
#include <hip/hip_runtime.h>

// ---------------- problem constants ----------------
#define TTOK   2048   // B*S
#define BATCH  2
#define SEQ    1024
#define HDIM   2048
#define NHEADS 32
#define NKVH   8
#define HDHEAD 64
#define NEXP   16
#define EDIM   1024
#define TOPKN  4
#define WINSZ  128
#define LIMITV 7.0f
#define ALPHAV 1.702f
#define EPSV   1e-6f
#define SCALEV 0.125f   // 64^-0.5

#define LDSSTRIDE 40          // padded column stride (elems); 80B, 16B-aligned
#define LDSTILE   (64 * LDSSTRIDE)

// ---------------- WMMA types & helpers ----------------
typedef __attribute__((ext_vector_type(16))) __bf16 v16bf;
typedef __attribute__((ext_vector_type(8)))  float  v8f;

union FragU {
    v16bf v;
    uint4 q[2];
    unsigned int u[8];
};

__device__ __forceinline__ unsigned short f2bf(float f) {
    unsigned int x = __float_as_uint(f);
    unsigned int r = x + 0x7fffu + ((x >> 16) & 1u);
    return (unsigned short)(r >> 16);
}
__device__ __forceinline__ float bf2f(unsigned short h) {
    return __uint_as_float(((unsigned int)h) << 16);
}

__device__ __forceinline__ v8f wmma_bf16(v16bf a, v16bf b, v8f c) {
    return __builtin_amdgcn_wmma_f32_16x16x32_bf16(false, a, false, b, (short)0, c,
                                                   false, false);
}

// Async 16B copy global -> LDS (per lane), tracked by ASYNCcnt.
__device__ __forceinline__ void async_g2l_b128(unsigned int lds_addr,
                                               const void* gaddr) {
    asm volatile("global_load_async_to_lds_b128 %0, %1, off"
                 :: "v"(lds_addr), "v"((unsigned long long)(size_t)gaddr)
                 : "memory");
}
__device__ __forceinline__ void wait_async0() {
    asm volatile("s_wait_asynccnt 0x0" ::: "memory");
}

// ---------------- shared GEMM core: async double-buffered LDS B tile --------
// Block = (32,8). Per k-step, 256 threads async-stage a 64(N) x 32(K) bf16
// tile of W'[N][K] directly into the ping-pong LDS buffer (one async b128
// each) while WMMAs consume the other buffer. A fragments are prefetched one
// step ahead in registers. LDS layout: (col, k) at lds[col*LDSSTRIDE + k].
__device__ __forceinline__ void gemm_core_lds(const unsigned short* __restrict__ arow,
                                              bool lane_valid,
                                              const unsigned short* __restrict__ W,
                                              int ldk, int K, int nb,
                                              int lane, int tid, bool wave_active,
                                              unsigned short* lds, v8f acc[4]) {
    int col = tid >> 2;
    int kp  = (tid & 3) << 3;
    const unsigned short* gsrc = W + (size_t)(nb + col) * ldk + kp;
    // low 32 bits of a generic shared pointer == wave-relative LDS byte offset
    unsigned int lslot =
        (unsigned int)(size_t)(void*)(lds + col * LDSSTRIDE + kp);
    int kh = (lane >> 4) << 3;
    const unsigned short* asrc0 = arow + kh;
    const unsigned short* asrc1 = arow + kh + 16;

    // prologue: async-stage tile 0 into buffer 0
    async_g2l_b128(lslot, gsrc);
    FragU af;
    if (lane_valid) {
        af.q[0] = *(const uint4*)asrc0;
        af.q[1] = *(const uint4*)asrc1;
    } else {
        af.q[0] = make_uint4(0u, 0u, 0u, 0u);
        af.q[1] = make_uint4(0u, 0u, 0u, 0u);
    }
    wait_async0();
    __syncthreads();

    int cur = 0;
    for (int kk = 0; kk < K; kk += 32) {
        int kn = kk + 32;
        bool more = kn < K;
        FragU anf;
        anf.q[0] = make_uint4(0u, 0u, 0u, 0u);
        anf.q[1] = anf.q[0];
        if (more) {
            // issue next-tile async copy + A prefetch before this tile's WMMAs
            async_g2l_b128(lslot + (unsigned int)((1 - cur) * LDSTILE * 2),
                           gsrc + kn);
            if (lane_valid) {
                anf.q[0] = *(const uint4*)(asrc0 + kn);
                anf.q[1] = *(const uint4*)(asrc1 + kn);
            }
        }
        if (wave_active) {
            const unsigned short* base = lds + cur * LDSTILE;
            FragU bfr[4];
#pragma unroll
            for (int nt = 0; nt < 4; ++nt) {
                const unsigned short* p =
                    base + (nt * 16 + (lane & 15)) * LDSSTRIDE + ((lane >> 4) << 3);
                bfr[nt].q[0] = ((const uint4*)p)[0];
                bfr[nt].q[1] = ((const uint4*)p)[2];
            }
#pragma unroll
            for (int nt = 0; nt < 4; ++nt)
                acc[nt] = wmma_bf16(af.v, bfr[nt].v, acc[nt]);
        }
        if (more) {
            wait_async0();
            __syncthreads();
            cur ^= 1;
            af = anf;
        }
    }
}

// ---------------- flat fp32 -> bf16 convert (weights, [N,K] kept) -----------
__global__ __launch_bounds__(256) void k_convert_bf16(const float* __restrict__ src,
                                                      unsigned short* __restrict__ dst,
                                                      int n4) {   // count/4
    int i = blockIdx.x * 256 + threadIdx.x;
    if (i >= n4) return;
    float4 t = ((const float4*)src)[i];
    ushort4 o;
    o.x = f2bf(t.x); o.y = f2bf(t.y); o.z = f2bf(t.z); o.w = f2bf(t.w);
    ((ushort4*)dst)[i] = o;
}

// ---------------- RMSNorm (fp32 in -> bf16 out) ----------------
__global__ __launch_bounds__(256) void k_rmsnorm(const float* __restrict__ x,
                                                 const float* __restrict__ wgt,
                                                 unsigned short* __restrict__ out) {
    __shared__ float red[256];
    int t = blockIdx.x;
    const float* row = x + (size_t)t * HDIM;
    float s = 0.f;
    for (int i = threadIdx.x; i < HDIM; i += 256) { float v = row[i]; s += v * v; }
    red[threadIdx.x] = s;
    __syncthreads();
    for (int o = 128; o > 0; o >>= 1) {
        if (threadIdx.x < o) red[threadIdx.x] += red[threadIdx.x + o];
        __syncthreads();
    }
    float r = rsqrtf(red[0] * (1.0f / HDIM) + EPSV);
    for (int i = threadIdx.x; i < HDIM; i += 256)
        out[(size_t)t * HDIM + i] = f2bf(row[i] * r * wgt[i]);
}

// ---------------- dense GEMM: C[M,N] = A_bf16[M,K] @ Wbf16[N,K]^T + bias (+resid)
// grid=(N/64, M/128), block=(32,8)
__global__ __launch_bounds__(256) void k_gemm_xw(const unsigned short* __restrict__ A,
                                                 const unsigned short* __restrict__ W,
                                                 const float* __restrict__ bias,
                                                 const float* __restrict__ resid,
                                                 float* __restrict__ C,
                                                 int M, int N, int K) {
    __shared__ unsigned short lds[2 * LDSTILE];
    int lane = threadIdx.x, wid = threadIdx.y;
    int tid = wid * 32 + lane;
    int m0 = (blockIdx.y * 8 + wid) * 16;
    int nb = blockIdx.x * 64;
    const unsigned short* arow = A + (size_t)(m0 + (lane & 15)) * K;
    v8f acc[4] = {};
    gemm_core_lds(arow, true, W, K, K, nb, lane, tid, true, lds, acc);
    int rbase = m0 + ((lane >> 4) << 3);
#pragma unroll
    for (int nt = 0; nt < 4; ++nt) {
        int col = nb + nt * 16 + (lane & 15);
        float bv = bias[col];
#pragma unroll
        for (int v = 0; v < 8; ++v) {
            size_t idx = (size_t)(rbase + v) * N + col;
            float val = acc[nt][v] + bv;
            if (resid) val += resid[idx];
            C[idx] = val;
        }
    }
}

// ---------------- RoPE (in-place on q and k) ----------------
__global__ __launch_bounds__(256) void k_rope(float* __restrict__ q,
                                              float* __restrict__ k,
                                              const float* __restrict__ cosb,
                                              const float* __restrict__ sinb) {
    int gid = blockIdx.x * 256 + threadIdx.x;
    int i = gid & 31;
    int rest = gid >> 5;
    int hh = rest % (NHEADS + NKVH);
    int t = rest / (NHEADS + NKVH);
    float* p;
    if (hh < NHEADS) p = q + ((size_t)t * NHEADS + hh) * HDHEAD;
    else             p = k + ((size_t)t * NKVH + (hh - NHEADS)) * HDHEAD;
    float c1 = cosb[(size_t)t * HDHEAD + i];
    float c2 = cosb[(size_t)t * HDHEAD + i + 32];
    float s1 = sinb[(size_t)t * HDHEAD + i];
    float s2 = sinb[(size_t)t * HDHEAD + i + 32];
    float x1 = p[i], x2 = p[i + 32];
    p[i]      = x1 * c1 - x2 * s1;
    p[i + 32] = x2 * c2 + x1 * s2;
}

// ---------------- sliding-window attention with sink (thread per (t, head)) --
__global__ __launch_bounds__(256) void k_attn(const float* __restrict__ q,
                                              const float* __restrict__ k,
                                              const float* __restrict__ v,
                                              const float* __restrict__ sinks,
                                              unsigned short* __restrict__ out) {
    int gid = blockIdx.x * 256 + threadIdx.x;
    int h = gid & (NHEADS - 1);
    int t = gid >> 5;
    int s = t & (SEQ - 1);
    int b = t >> 10;
    int kvh = h >> 2;  // NH/NKV = 4
    float4 qv[16];
    const float4* qp = (const float4*)(q + ((size_t)t * NHEADS + h) * HDHEAD);
#pragma unroll
    for (int i = 0; i < 16; ++i) qv[i] = qp[i];
    float m = sinks[h];
    float l = 1.0f;           // sink contributes exp(sink - m) = 1 at init
    float4 acc[16] = {};
    int j0 = s - (WINSZ - 1); if (j0 < 0) j0 = 0;
    for (int j = j0; j <= s; ++j) {
        const float4* kp =
            (const float4*)(k + (((size_t)(b * SEQ + j)) * NKVH + kvh) * HDHEAD);
        float dot = 0.f;
#pragma unroll
        for (int i = 0; i < 16; ++i) {
            float4 kv4 = kp[i];
            dot += qv[i].x * kv4.x + qv[i].y * kv4.y + qv[i].z * kv4.z + qv[i].w * kv4.w;
        }
        dot *= SCALEV;
        float mn = fmaxf(m, dot);
        float sc = __expf(m - mn);
        float p  = __expf(dot - mn);
        l = l * sc + p;
        const float4* vp =
            (const float4*)(v + (((size_t)(b * SEQ + j)) * NKVH + kvh) * HDHEAD);
#pragma unroll
        for (int i = 0; i < 16; ++i) {
            float4 vv = vp[i];
            acc[i].x = acc[i].x * sc + p * vv.x;
            acc[i].y = acc[i].y * sc + p * vv.y;
            acc[i].z = acc[i].z * sc + p * vv.z;
            acc[i].w = acc[i].w * sc + p * vv.w;
        }
        m = mn;
    }
    float inv = 1.0f / l;
    unsigned short* op = out + ((size_t)t * NHEADS + h) * HDHEAD;
#pragma unroll
    for (int i = 0; i < 16; ++i) {
        op[4 * i + 0] = f2bf(acc[i].x * inv);
        op[4 * i + 1] = f2bf(acc[i].y * inv);
        op[4 * i + 2] = f2bf(acc[i].z * inv);
        op[4 * i + 3] = f2bf(acc[i].w * inv);
    }
}

// ---------------- router logits: one block per token, 8 waves x 2 experts ----
__global__ __launch_bounds__(256) void k_router(const unsigned short* __restrict__ xn,
                                                const float* __restrict__ rw,
                                                const float* __restrict__ rb,
                                                float* __restrict__ logits) {
    int t = blockIdx.x;
    int lane = threadIdx.x, wid = threadIdx.y;
#pragma unroll
    for (int rep = 0; rep < 2; ++rep) {
        int e = wid * 2 + rep;
        const float* wrow = rw + (size_t)e * HDIM;
        float s = 0.f;
        for (int i = lane; i < HDIM; i += 32)
            s += bf2f(xn[(size_t)t * HDIM + i]) * wrow[i];
        for (int o = 16; o > 0; o >>= 1) s += __shfl_xor(s, o, 32);
        if (lane == 0) logits[t * NEXP + e] = s + rb[e];
    }
}

// ---------------- top-4 + softmax + expert counting ----------------
__global__ __launch_bounds__(256) void k_topk(const float* __restrict__ logits,
                                              float* __restrict__ topw,
                                              int* __restrict__ topi,
                                              int* __restrict__ counts) {
    int t = blockIdx.x * 256 + threadIdx.x;
    if (t >= TTOK) return;
    float lg[NEXP];
#pragma unroll
    for (int e = 0; e < NEXP; ++e) lg[e] = logits[t * NEXP + e];
    float tv[TOPKN]; int ti[TOPKN];
#pragma unroll
    for (int kk = 0; kk < TOPKN; ++kk) {
        float best = -3.4e38f; int bi = 0;
#pragma unroll
        for (int e = 0; e < NEXP; ++e)
            if (lg[e] > best) { best = lg[e]; bi = e; }
        tv[kk] = best; ti[kk] = bi; lg[bi] = -3.4e38f;
    }
    float mx = tv[0], sum = 0.f, pw[TOPKN];
#pragma unroll
    for (int kk = 0; kk < TOPKN; ++kk) { pw[kk] = __expf(tv[kk] - mx); sum += pw[kk]; }
    float inv = 1.0f / sum;
#pragma unroll
    for (int kk = 0; kk < TOPKN; ++kk) {
        topw[t * TOPKN + kk] = pw[kk] * inv;
        topi[t * TOPKN + kk] = ti[kk];
        atomicAdd(&counts[ti[kk]], 1);
    }
}

__global__ void k_scan(const int* __restrict__ counts, int* __restrict__ offsets) {
    if (blockIdx.x == 0 && threadIdx.x == 0) {
        int a = 0;
        for (int e = 0; e < NEXP; ++e) { offsets[e] = a; a += counts[e]; }
    }
}

__global__ __launch_bounds__(256) void k_assign(const int* __restrict__ topi,
                                                const int* __restrict__ offsets,
                                                int* __restrict__ fill,
                                                int* __restrict__ token_list,
                                                int* __restrict__ slot_of) {
    int idx = blockIdx.x * 256 + threadIdx.x;
    if (idx >= TTOK * TOPKN) return;
    int e = topi[idx];
    int pos = offsets[e] + atomicAdd(&fill[e], 1);
    token_list[pos] = idx >> 2;
    slot_of[idx] = pos;
}

// ---------------- tiled transpose + fp32->bf16: in[z][R][C] -> out[z][C][R] ---
__global__ __launch_bounds__(256) void k_transpose_cvt(const float* __restrict__ in,
                                                       unsigned short* __restrict__ out,
                                                       int R, int C) {
    __shared__ float tile[32][33];
    size_t base = (size_t)blockIdx.z * R * C;
    int c0 = blockIdx.x * 32, r0 = blockIdx.y * 32;
    int tx = threadIdx.x, ty = threadIdx.y;
#pragma unroll
    for (int j = 0; j < 32; j += 8)
        tile[ty + j][tx] = in[base + (size_t)(r0 + ty + j) * C + c0 + tx];
    __syncthreads();
#pragma unroll
    for (int j = 0; j < 32; j += 8)
        out[base + (size_t)(c0 + ty + j) * R + r0 + tx] = f2bf(tile[tx][ty + j]);
}

// ---------------- MoE gate_up GEMM + clamp/SiLU epilogue --------------------
// grid = (2ED/64, 16, E); block = (32,8). gated[slot][ED] bf16 out.
__global__ __launch_bounds__(256) void k_moe_gateup(const unsigned short* __restrict__ xn,
                                                    const unsigned short* __restrict__ wgu,
                                                    const float* __restrict__ gub,
                                                    const int* __restrict__ token_list,
                                                    const int* __restrict__ offsets,
                                                    const int* __restrict__ counts,
                                                    unsigned short* __restrict__ gated) {
    __shared__ unsigned short lds[2 * LDSTILE];
    int e = blockIdx.z;
    int cnt = counts[e];
    int mblk = blockIdx.y * 128;
    if (mblk >= cnt) return;                 // uniform across block
    int lane = threadIdx.x, wid = threadIdx.y;
    int tid = wid * 32 + lane;
    int m0 = mblk + wid * 16;
    bool wave_active = m0 < cnt;
    int off = offsets[e];
    int nb = blockIdx.x * 64;
    int mrow = m0 + (lane & 15);
    bool valid = wave_active && (mrow < cnt);
    int tok = valid ? token_list[off + mrow] : 0;
    const unsigned short* arow = xn + (size_t)tok * HDIM;
    const unsigned short* wbase = wgu + (size_t)e * (2 * EDIM) * HDIM;
    v8f acc[4] = {};
    gemm_core_lds(arow, valid, wbase, HDIM, HDIM, nb, lane, tid, wave_active, lds, acc);
    if (!wave_active) return;
    int rbase = m0 + ((lane >> 4) << 3);
#pragma unroll
    for (int nt = 0; nt < 4; ++nt) {
        int c = nb + nt * 16 + (lane & 15);
        float bv = gub[e * 2 * EDIM + c];
#pragma unroll
        for (int v = 0; v < 8; ++v) {
            float gv = acc[nt][v] + bv;
            float other = __shfl_xor(gv, 1, 32);  // pair interleaved gate/up cols
            if ((c & 1) == 0) {
                float gate = fminf(gv, LIMITV);
                float up = fminf(fmaxf(other, -LIMITV), LIMITV);
                float act = (up + 1.0f) * (gate / (1.0f + __expf(-gate * ALPHAV)));
                int mm = rbase + v;
                if (mm < cnt)
                    gated[(size_t)(off + mm) * EDIM + (c >> 1)] = f2bf(act);
            }
        }
    }
}

// ---------------- MoE down GEMM: slot_out[slot][H] = gated @ wdn^T + bias ----
__global__ __launch_bounds__(256) void k_moe_down(const unsigned short* __restrict__ gated,
                                                  const unsigned short* __restrict__ wdn,
                                                  const float* __restrict__ db,
                                                  const int* __restrict__ offsets,
                                                  const int* __restrict__ counts,
                                                  float* __restrict__ slot_out) {
    __shared__ unsigned short lds[2 * LDSTILE];
    int e = blockIdx.z;
    int cnt = counts[e];
    int mblk = blockIdx.y * 128;
    if (mblk >= cnt) return;                 // uniform across block
    int lane = threadIdx.x, wid = threadIdx.y;
    int tid = wid * 32 + lane;
    int m0 = mblk + wid * 16;
    bool wave_active = m0 < cnt;
    int off = offsets[e];
    int nb = blockIdx.x * 64;
    int mrow = m0 + (lane & 15);
    bool valid = wave_active && (mrow < cnt);
    const unsigned short* arow = gated + (size_t)(off + (valid ? mrow : 0)) * EDIM;
    const unsigned short* wbase = wdn + (size_t)e * HDIM * EDIM;
    v8f acc[4] = {};
    gemm_core_lds(arow, valid, wbase, EDIM, EDIM, nb, lane, tid, wave_active, lds, acc);
    if (!wave_active) return;
    int rbase = m0 + ((lane >> 4) << 3);
#pragma unroll
    for (int nt = 0; nt < 4; ++nt) {
        int c = nb + nt * 16 + (lane & 15);
        float bv = db[e * HDIM + c];
#pragma unroll
        for (int v = 0; v < 8; ++v) {
            int mm = rbase + v;
            if (mm < cnt)
                slot_out[(size_t)(off + mm) * HDIM + c] = acc[nt][v] + bv;
        }
    }
}

// ---------------- final gather: residual + sum_k w_k * expert_out -----------
__global__ __launch_bounds__(256) void k_gather(const float* __restrict__ hidden2,
                                                const float* __restrict__ slot_out,
                                                const float* __restrict__ topw,
                                                const int* __restrict__ slot_of,
                                                float* __restrict__ out) {
    size_t gid = (size_t)blockIdx.x * 256 + threadIdx.x;
    int t = (int)(gid >> 11);      // / HDIM
    int hcol = (int)(gid & (HDIM - 1));
    float s = hidden2[gid];
#pragma unroll
    for (int kk = 0; kk < TOPKN; ++kk) {
        int slot = slot_of[t * TOPKN + kk];
        s += topw[t * TOPKN + kk] * slot_out[(size_t)slot * HDIM + hcol];
    }
    out[gid] = s;
}

// ---------------- host launcher ----------------
extern "C" void kernel_launch(void* const* d_in, const int* in_sizes, int n_in,
                              void* d_out, int out_size, void* d_ws, size_t ws_size,
                              hipStream_t stream) {
    const float* hidden = (const float*)d_in[0];
    const float* cosb   = (const float*)d_in[1];
    const float* sinb   = (const float*)d_in[2];
    const float* ln1w   = (const float*)d_in[3];
    const float* ln2w   = (const float*)d_in[4];
    const float* qw     = (const float*)d_in[5];
    const float* qb     = (const float*)d_in[6];
    const float* kw     = (const float*)d_in[7];
    const float* kbi    = (const float*)d_in[8];
    const float* vw     = (const float*)d_in[9];
    const float* vb     = (const float*)d_in[10];
    const float* ow     = (const float*)d_in[11];
    const float* ob     = (const float*)d_in[12];
    const float* sinks  = (const float*)d_in[13];
    const float* rw     = (const float*)d_in[14];
    const float* rb     = (const float*)d_in[15];
    const float* gup    = (const float*)d_in[16];
    const float* gub    = (const float*)d_in[17];
    const float* dnp    = (const float*)d_in[18];
    const float* dnb    = (const float*)d_in[19];
    float* out = (float*)d_out;

    char* w = (char*)d_ws;
    size_t cursor = 0;
    auto alloc = [&](size_t bytes) {
        void* p = w + cursor;
        cursor = (cursor + bytes + 255) & ~(size_t)255;
        return p;
    };
    unsigned short* xn1   = (unsigned short*)alloc((size_t)TTOK * HDIM * 2);
    float* qbuf           = (float*)alloc((size_t)TTOK * NHEADS * HDHEAD * 4);
    float* kbuf           = (float*)alloc((size_t)TTOK * NKVH * HDHEAD * 4);
    float* vbuf           = (float*)alloc((size_t)TTOK * NKVH * HDHEAD * 4);
    unsigned short* attnb = (unsigned short*)alloc((size_t)TTOK * HDIM * 2);
    float* hidden2        = (float*)alloc((size_t)TTOK * HDIM * 4);
    unsigned short* xn2   = (unsigned short*)alloc((size_t)TTOK * HDIM * 2);
    float* logits         = (float*)alloc((size_t)TTOK * NEXP * 4);
    float* topw           = (float*)alloc((size_t)TTOK * TOPKN * 4);
    int* topi             = (int*)alloc((size_t)TTOK * TOPKN * 4);
    int* ctrs             = (int*)alloc(3 * NEXP * 4);  // counts | fill | offsets
    int* token_list       = (int*)alloc((size_t)TTOK * TOPKN * 4);
    int* slot_of          = (int*)alloc((size_t)TTOK * TOPKN * 4);
    unsigned short* qwb   = (unsigned short*)alloc((size_t)2048 * HDIM * 2);
    unsigned short* kwb   = (unsigned short*)alloc((size_t)512 * HDIM * 2);
    unsigned short* vwb   = (unsigned short*)alloc((size_t)512 * HDIM * 2);
    unsigned short* owb   = (unsigned short*)alloc((size_t)HDIM * HDIM * 2);
    unsigned short* wgu   = (unsigned short*)alloc((size_t)NEXP * 2 * EDIM * HDIM * 2);
    unsigned short* wdn   = (unsigned short*)alloc((size_t)NEXP * HDIM * EDIM * 2);
    unsigned short* gated = (unsigned short*)alloc((size_t)TTOK * TOPKN * EDIM * 2);
    float* slot_out       = (float*)alloc((size_t)TTOK * TOPKN * HDIM * 4);

    int* counts  = ctrs;
    int* fill    = ctrs + NEXP;
    int* offsets = ctrs + 2 * NEXP;

    dim3 blkW(32, 8, 1);
    dim3 blk256(256, 1, 1);

    hipMemsetAsync(ctrs, 0, 3 * NEXP * 4, stream);

    // one-time-per-call weight conversions (independent of the attention chain)
    k_convert_bf16<<<(2048 * HDIM / 4 + 255) / 256, blk256, 0, stream>>>(
        qw, qwb, 2048 * HDIM / 4);
    k_convert_bf16<<<(512 * HDIM / 4 + 255) / 256, blk256, 0, stream>>>(
        kw, kwb, 512 * HDIM / 4);
    k_convert_bf16<<<(512 * HDIM / 4 + 255) / 256, blk256, 0, stream>>>(
        vw, vwb, 512 * HDIM / 4);
    k_convert_bf16<<<(HDIM * HDIM / 4 + 255) / 256, blk256, 0, stream>>>(
        ow, owb, HDIM * HDIM / 4);
    k_transpose_cvt<<<dim3(2 * EDIM / 32, HDIM / 32, NEXP), blkW, 0, stream>>>(
        gup, wgu, HDIM, 2 * EDIM);
    k_transpose_cvt<<<dim3(HDIM / 32, EDIM / 32, NEXP), blkW, 0, stream>>>(
        dnp, wdn, EDIM, HDIM);

    // attention block
    k_rmsnorm<<<TTOK, blk256, 0, stream>>>(hidden, ln1w, xn1);
    k_gemm_xw<<<dim3(2048 / 64, TTOK / 128, 1), blkW, 0, stream>>>(
        xn1, qwb, qb, nullptr, qbuf, TTOK, 2048, HDIM);
    k_gemm_xw<<<dim3(512 / 64, TTOK / 128, 1), blkW, 0, stream>>>(
        xn1, kwb, kbi, nullptr, kbuf, TTOK, 512, HDIM);
    k_gemm_xw<<<dim3(512 / 64, TTOK / 128, 1), blkW, 0, stream>>>(
        xn1, vwb, vb, nullptr, vbuf, TTOK, 512, HDIM);
    k_rope<<<(TTOK * (NHEADS + NKVH) * 32) / 256, blk256, 0, stream>>>(
        qbuf, kbuf, cosb, sinb);
    k_attn<<<(TTOK * NHEADS) / 256, blk256, 0, stream>>>(
        qbuf, kbuf, vbuf, sinks, attnb);
    k_gemm_xw<<<dim3(HDIM / 64, TTOK / 128, 1), blkW, 0, stream>>>(
        attnb, owb, ob, hidden, hidden2, TTOK, HDIM, HDIM);

    // MoE block
    k_rmsnorm<<<TTOK, blk256, 0, stream>>>(hidden2, ln2w, xn2);
    k_router<<<TTOK, blkW, 0, stream>>>(xn2, rw, rb, logits);
    k_topk<<<TTOK / 256, blk256, 0, stream>>>(logits, topw, topi, counts);
    k_scan<<<1, 1, 0, stream>>>(counts, offsets);
    k_assign<<<(TTOK * TOPKN) / 256, blk256, 0, stream>>>(
        topi, offsets, fill, token_list, slot_of);
    k_moe_gateup<<<dim3(2 * EDIM / 64, 16, NEXP), blkW, 0, stream>>>(
        xn2, wgu, gub, token_list, offsets, counts, gated);
    k_moe_down<<<dim3(HDIM / 64, 16, NEXP), blkW, 0, stream>>>(
        gated, wdn, dnb, offsets, counts, slot_out);
    k_gather<<<(TTOK * HDIM) / 256, blk256, 0, stream>>>(
        hidden2, slot_out, topw, slot_of, out);
}
